// VariationalQuantumClassifier_85830626443684
// MI455X (gfx1250) — matrix operand order repaired and checked by reading, not compiled, not run on hardware
//
#include <hip/hip_runtime.h>

// ---------------------------------------------------------------------------
// VariationalQuantumClassifier for MI455X (gfx1250).
//
//   1) split_bf16_kernel     : f32 -> (bf16_hi, bf16_lo) planes for x, W_red
//   2) gemm_split_lds_kernel : xr = x @ W_red^T + b_red via
//                              v_wmma_f32_16x16x32_bf16, 3-product split
//                              (hi*hi + hi*lo + lo*hi). Operands staged with
//                              GLOBAL_LOAD_ASYNC_TO_LDS_B128, double-buffered
//                              in LDS, consumed via ds_load_b128.
//   3) quantum_kernel        : per-row 12-qubit circuit in LDS
//   4) mlp_kernel            : 12->32->16->1 MLP + sigmoid
// ---------------------------------------------------------------------------

#define N_QUBITS  12
#define STATE_DIM 4096
#define BATCH     8192
#define FDIM      8192
#define N_LAYERS  3

typedef __attribute__((ext_vector_type(16))) __bf16 v16bf;
typedef __attribute__((ext_vector_type(8)))  float  v8f;

__device__ __forceinline__ unsigned short bf16_rne(float f) {
    unsigned int u = __builtin_bit_cast(unsigned int, f);
    unsigned int r = u + 0x7FFFu + ((u >> 16) & 1u);   // round-to-nearest-even
    return (unsigned short)(r >> 16);
}
__device__ __forceinline__ float bf16_to_f32(unsigned short h) {
    return __builtin_bit_cast(float, (unsigned int)h << 16);
}

// --------------------------- async copy helpers ----------------------------
// Builtin signature (from hipcc diagnostic): param0 is a non-const
//   'int __vector_size__(16) __device__ *'  == v4i addrspace(1)*   (global src)
// param1 is the LDS destination (addrspace(3)), then imm offset + cpol.
typedef int v4i __attribute__((vector_size(16)));
typedef __attribute__((address_space(1))) v4i* gv4ip;
typedef __attribute__((address_space(3))) v4i* lv4ip;

__device__ __forceinline__ void async_copy_b128(const char* g, char* l) {
#if __has_builtin(__builtin_amdgcn_global_load_async_to_lds_b128)
    __builtin_amdgcn_global_load_async_to_lds_b128((gv4ip)g, (lv4ip)l, 0, 0);
#else
    unsigned lv = (unsigned)(unsigned long long)(
        (__attribute__((address_space(3))) char*)l);
    asm volatile("global_load_async_to_lds_b128 %0, %1, off"
                 :: "v"(lv), "v"(g) : "memory");
#endif
}

template <int N>
__device__ __forceinline__ void wait_asynccnt() {
#if __has_builtin(__builtin_amdgcn_s_wait_asynccnt)
    __builtin_amdgcn_s_wait_asynccnt(N);
#else
    asm volatile("s_wait_asynccnt %0" :: "i"(N) : "memory");
#endif
}

// ---------------------------------------------------------------------------
// 1) Split f32 into bf16 hi/lo planes (4 elements per thread, vectorized I/O)
// ---------------------------------------------------------------------------
__global__ __launch_bounds__(256)
void split_bf16_kernel(const float* __restrict__ src,
                       unsigned short* __restrict__ hi,
                       unsigned short* __restrict__ lo,
                       long long n4) {
    long long t = (long long)blockIdx.x * blockDim.x + threadIdx.x;
    if (t >= n4) return;
    long long base = t * 4;
    float4 v = *(const float4*)(src + base);

    unsigned short h0 = bf16_rne(v.x), h1 = bf16_rne(v.y);
    unsigned short h2 = bf16_rne(v.z), h3 = bf16_rne(v.w);
    unsigned short l0 = bf16_rne(v.x - bf16_to_f32(h0));
    unsigned short l1 = bf16_rne(v.y - bf16_to_f32(h1));
    unsigned short l2 = bf16_rne(v.z - bf16_to_f32(h2));
    unsigned short l3 = bf16_rne(v.w - bf16_to_f32(h3));

    uint2 hp, lp;
    hp.x = (unsigned)h0 | ((unsigned)h1 << 16);
    hp.y = (unsigned)h2 | ((unsigned)h3 << 16);
    lp.x = (unsigned)l0 | ((unsigned)l1 << 16);
    lp.y = (unsigned)l2 | ((unsigned)l3 << 16);
    *(uint2*)(hi + base) = hp;
    *(uint2*)(lo + base) = lp;
}

// ---------------------------------------------------------------------------
// 2) Split-bf16 WMMA GEMM with async-LDS staging.
//    Block = 128x128 tile, 8 waves (2x4), wave = 64x32 = 4x2 WMMA subtiles.
//    LDS: 2 buffers x 4 planes (Ahi, Alo, Bhi, Blo) x 128 rows x 80B (padded;
//    bank-conflict-free fragment reads since gcd(20,64)=4 covers 16 lanes).
// ---------------------------------------------------------------------------
#define KSTEP          32
#define LDS_ROW_BYTES  80                       // 32 bf16 = 64B data + 16B pad
#define PLANE_BYTES    (128 * LDS_ROW_BYTES)    // 10240
#define BUF_BYTES      (4 * PLANE_BYTES)        // 40960
#define OFF_AHI        0
#define OFF_ALO        (1 * PLANE_BYTES)
#define OFF_BHI        (2 * PLANE_BYTES)
#define OFF_BLO        (3 * PLANE_BYTES)
#define GEMM_LDS_BYTES (2 * BUF_BYTES)          // 81920

union FragAB { v16bf v; uint4 q[2]; };

__global__ __launch_bounds__(256)
void gemm_split_lds_kernel(const unsigned short* __restrict__ xHi,
                           const unsigned short* __restrict__ xLo,
                           const unsigned short* __restrict__ wHi,
                           const unsigned short* __restrict__ wLo,
                           const float* __restrict__ bred,
                           float* __restrict__ xr) {
    extern __shared__ char lds[];

    const int tid   = threadIdx.x;
    const int lane  = tid & 31;
    const int wave  = tid >> 5;
    const int wm    = wave >> 2;        // 0..1  (M direction)
    const int wn    = wave & 3;         // 0..3  (N direction)
    const int l16   = lane & 15;
    const int lhalf = lane >> 4;

    const int mBB = blockIdx.y * 128;   // block M base
    const int nBB = blockIdx.x * 128;   // block N base

    // ---- async staging: each thread copies 8 x 16B per stage --------------
    const int c4  = tid & 3;            // which 16B chunk of the 64B row
    const int r0  = tid >> 2;           // 0..63 (second row = r0+64)
    const char* gAhi = (const char*)xHi;
    const char* gAlo = (const char*)xLo;
    const char* gBhi = (const char*)wHi;
    const char* gBlo = (const char*)wLo;

    auto issue_stage = [&](int bufOff, int k) {
#pragma unroll
        for (int t = 0; t < 2; ++t) {
            int r    = r0 + t * 64;
            int lOff = bufOff + r * LDS_ROW_BYTES + c4 * 16;
            long long gA = ((long long)(mBB + r) * FDIM + k + c4 * 8) * 2;
            long long gB = ((long long)(nBB + r) * FDIM + k + c4 * 8) * 2;
            async_copy_b128(gAhi + gA, lds + lOff + OFF_AHI);
            async_copy_b128(gAlo + gA, lds + lOff + OFF_ALO);
            async_copy_b128(gBhi + gB, lds + lOff + OFF_BHI);
            async_copy_b128(gBlo + gB, lds + lOff + OFF_BLO);
        }
    };

    // C/D layout: N = lane%16, M = vgpr_idx + 8*(lane/16); b_red depends on N.
    v8f acc[4][2];
    for (int ni = 0; ni < 2; ++ni) {
        float bv = bred[nBB + wn * 32 + ni * 16 + l16];
        for (int mi = 0; mi < 4; ++mi)
            for (int r = 0; r < 8; ++r) acc[mi][ni][r] = bv;
    }

    // Fragment LDS byte offsets (per lane, within a buffer).
    // 16-bit A layout: lane holds K chunks [kA,kA+8) and [kA+16,kA+24), kA=8*lhalf
    // 16-bit B layout: lane holds K chunks [kB,kB+8) and [kB+16,kB+24), kB=8*lhalf
    int aOff[4], bOff[2];
    for (int mi = 0; mi < 4; ++mi)
        aOff[mi] = (wm * 64 + mi * 16 + l16) * LDS_ROW_BYTES + lhalf * 16;
    for (int ni = 0; ni < 2; ++ni)
        bOff[ni] = (wn * 32 + ni * 16 + l16) * LDS_ROW_BYTES + lhalf * 16;

    // ---- pipelined main loop ---------------------------------------------
    issue_stage(0, 0);
    int buf = 0;
    for (int k = 0; k < FDIM; k += KSTEP) {
        if (k + KSTEP < FDIM) {
            issue_stage((buf ^ 1) * BUF_BYTES, k + KSTEP);
            wait_asynccnt<8>();     // in-order done => current stage landed
        } else {
            wait_asynccnt<0>();
        }
        __syncthreads();            // all waves see the staged tile

        const char* base = lds + buf * BUF_BYTES;
        FragAB aHi[4], aLo[4], bHi[2], bLo[2];
#pragma unroll
        for (int mi = 0; mi < 4; ++mi) {
            aHi[mi].q[0] = *(const uint4*)(base + OFF_AHI + aOff[mi]);
            aHi[mi].q[1] = *(const uint4*)(base + OFF_AHI + aOff[mi] + 32);
            aLo[mi].q[0] = *(const uint4*)(base + OFF_ALO + aOff[mi]);
            aLo[mi].q[1] = *(const uint4*)(base + OFF_ALO + aOff[mi] + 32);
        }
#pragma unroll
        for (int ni = 0; ni < 2; ++ni) {
            bHi[ni].q[0] = *(const uint4*)(base + OFF_BHI + bOff[ni]);
            bHi[ni].q[1] = *(const uint4*)(base + OFF_BHI + bOff[ni] + 32);
            bLo[ni].q[0] = *(const uint4*)(base + OFF_BLO + bOff[ni]);
            bLo[ni].q[1] = *(const uint4*)(base + OFF_BLO + bOff[ni] + 32);
        }
#pragma unroll
        for (int mi = 0; mi < 4; ++mi)
#pragma unroll
            for (int ni = 0; ni < 2; ++ni) {
                acc[mi][ni] = __builtin_amdgcn_wmma_f32_16x16x32_bf16(
                    false, aHi[mi].v, false, bHi[ni].v, (short)0, acc[mi][ni], false, false);
                acc[mi][ni] = __builtin_amdgcn_wmma_f32_16x16x32_bf16(
                    false, aHi[mi].v, false, bLo[ni].v, (short)0, acc[mi][ni], false, false);
                acc[mi][ni] = __builtin_amdgcn_wmma_f32_16x16x32_bf16(
                    false, aLo[mi].v, false, bHi[ni].v, (short)0, acc[mi][ni], false, false);
            }

        __syncthreads();            // done reading before buffer is rewritten
        buf ^= 1;
    }

    for (int mi = 0; mi < 4; ++mi)
        for (int ni = 0; ni < 2; ++ni) {
            int col     = nBB + wn * 32 + ni * 16 + l16;
            int rowBase = mBB + wm * 64 + mi * 16 + lhalf * 8;
            for (int r = 0; r < 8; ++r)
                xr[(long long)(rowBase + r) * STATE_DIM + col] = acc[mi][ni][r];
        }
}

// ---------------------------------------------------------------------------
// 3) Quantum circuit: one block per batch row; state in LDS.
// ---------------------------------------------------------------------------
__global__ __launch_bounds__(256)
void quantum_kernel(const float* __restrict__ xr,
                    const float* __restrict__ theta,
                    float* __restrict__ mfeat) {
    __shared__ float s[STATE_DIM];       // 16 KB
    __shared__ float red[256];
    __shared__ float pr[12 * 256];       // 12 KB

    const int tid = threadIdx.x;
    const long long row = blockIdx.x;
    const float* src = xr + row * STATE_DIM;

    float ss = 0.f;
    for (int e = 0; e < 16; ++e) {
        float v = src[e * 256 + tid];
        s[e * 256 + tid] = v;
        ss += v * v;
    }
    red[tid] = ss;
    __syncthreads();
    for (int off = 128; off > 0; off >>= 1) {
        if (tid < off) red[tid] += red[tid + off];
        __syncthreads();
    }
    float inv = 1.0f / (sqrtf(red[0]) + 1e-10f);
    for (int e = 0; e < 16; ++e) s[e * 256 + tid] *= inv;
    __syncthreads();

    for (int l = 0; l < N_LAYERS; ++l) {
        for (int q = 0; q < N_QUBITS; ++q) {   // RY butterflies on bit 11-q
            float th = theta[l * N_QUBITS + q] * 0.5f;
            float c = cosf(th), sn = sinf(th);
            int b = 11 - q, mask = 1 << b;
            for (int j = 0; j < 8; ++j) {
                int p  = j * 256 + tid;
                int i0 = ((p >> b) << (b + 1)) | (p & (mask - 1));
                int i1 = i0 | mask;
                float s0 = s[i0], s1 = s[i1];
                s[i0] = c * s0 - sn * s1;
                s[i1] = sn * s0 + c * s1;
            }
            __syncthreads();
        }
        for (int q = 0; q < N_QUBITS; ++q) {   // CNOT ring (involution swap)
            int t = (q + 1) % N_QUBITS;
            int pc = 11 - q, pt = 11 - t;
            int cmask = 1 << pc, tmask = 1 << pt;
            int pl = pc < pt ? pc : pt, ph = pc < pt ? pt : pc;
            int lmask = 1 << pl, hmask = 1 << ph;
            for (int j = 0; j < 4; ++j) {
                int p = j * 256 + tid;
                int i = ((p >> pl) << (pl + 1)) | (p & (lmask - 1));
                i = ((i >> ph) << (ph + 1)) | (i & (hmask - 1));
                i |= cmask;                    // cbit=1, tbit=0
                float a = s[i], b2 = s[i | tmask];
                s[i] = b2;
                s[i | tmask] = a;
            }
            __syncthreads();
        }
    }

    float accj[12];
    for (int j = 0; j < 12; ++j) accj[j] = 0.f;
    for (int e = 0; e < 16; ++e) {
        int i = e * 256 + tid;
        float p = s[i] * s[i];
        for (int j = 0; j < 12; ++j)
            accj[j] += ((i >> (11 - j)) & 1) ? -p : p;
    }
    for (int j = 0; j < 12; ++j) pr[j * 256 + tid] = accj[j];
    __syncthreads();
    for (int off = 128; off > 0; off >>= 1) {
        if (tid < off)
            for (int j = 0; j < 12; ++j)
                pr[j * 256 + tid] += pr[j * 256 + tid + off];
        __syncthreads();
    }
    if (tid < 12) mfeat[row * 12 + tid] = pr[tid * 256];
}

// ---------------------------------------------------------------------------
// 4) Tiny MLP head: 12 -> 32 -> 16 -> 1, sigmoid.  One thread per row.
// ---------------------------------------------------------------------------
__global__ __launch_bounds__(256)
void mlp_kernel(const float* __restrict__ mfeat,
                const float* __restrict__ W1, const float* __restrict__ b1,
                const float* __restrict__ W2, const float* __restrict__ b2,
                const float* __restrict__ W3, const float* __restrict__ b3,
                float* __restrict__ out) {
    int r = blockIdx.x * blockDim.x + threadIdx.x;
    if (r >= BATCH) return;
    float mv[12];
    for (int j = 0; j < 12; ++j) mv[j] = mfeat[r * 12 + j];
    float h1[32];
    for (int i = 0; i < 32; ++i) {
        float a = b1[i];
        for (int j = 0; j < 12; ++j) a += W1[i * 12 + j] * mv[j];
        h1[i] = fmaxf(a, 0.f);
    }
    float h2[16];
    for (int i = 0; i < 16; ++i) {
        float a = b2[i];
        for (int j = 0; j < 32; ++j) a += W2[i * 32 + j] * h1[j];
        h2[i] = fmaxf(a, 0.f);
    }
    float z = b3[0];
    for (int j = 0; j < 16; ++j) z += W3[j] * h2[j];
    out[r] = 1.f / (1.f + expf(-z));
}

// ---------------------------------------------------------------------------
extern "C" void kernel_launch(void* const* d_in, const int* in_sizes, int n_in,
                              void* d_out, int out_size, void* d_ws, size_t ws_size,
                              hipStream_t stream) {
    const float* x     = (const float*)d_in[0];
    const float* Wred  = (const float*)d_in[1];
    const float* bred  = (const float*)d_in[2];
    const float* theta = (const float*)d_in[3];
    const float* W1    = (const float*)d_in[4];
    const float* b1    = (const float*)d_in[5];
    const float* W2    = (const float*)d_in[6];
    const float* b2    = (const float*)d_in[7];
    const float* W3    = (const float*)d_in[8];
    const float* b3    = (const float*)d_in[9];

    char* ws = (char*)d_ws;
    size_t off = 0;
    auto carve = [&](size_t bytes) -> void* {
        void* p = ws + off;
        off += (bytes + 255) & ~(size_t)255;
        return p;
    };
    unsigned short* xHi = (unsigned short*)carve((size_t)BATCH * FDIM * 2);
    unsigned short* xLo = (unsigned short*)carve((size_t)BATCH * FDIM * 2);
    unsigned short* wHi = (unsigned short*)carve((size_t)STATE_DIM * FDIM * 2);
    unsigned short* wLo = (unsigned short*)carve((size_t)STATE_DIM * FDIM * 2);
    float* xr    = (float*)carve((size_t)BATCH * STATE_DIM * sizeof(float));
    float* mfeat = (float*)carve((size_t)BATCH * 12 * sizeof(float));

    {
        long long n4 = (long long)BATCH * FDIM / 4;
        split_bf16_kernel<<<(unsigned)((n4 + 255) / 256), 256, 0, stream>>>(x, xHi, xLo, n4);
    }
    {
        long long n4 = (long long)STATE_DIM * FDIM / 4;
        split_bf16_kernel<<<(unsigned)((n4 + 255) / 256), 256, 0, stream>>>(Wred, wHi, wLo, n4);
    }

    dim3 ggrid(STATE_DIM / 128, BATCH / 128);   // (N tiles, M tiles)
    gemm_split_lds_kernel<<<ggrid, 256, GEMM_LDS_BYTES, stream>>>(
        xHi, xLo, wHi, wLo, bred, xr);

    quantum_kernel<<<BATCH, 256, 0, stream>>>(xr, theta, mfeat);

    mlp_kernel<<<BATCH / 256, 256, 0, stream>>>(mfeat, W1, b1, W2, b2, W3, b3,
                                                (float*)d_out);
}
